// Net_55353538511450
// MI455X (gfx1250) — matrix operand order repaired and checked by reading, not compiled, not run on hardware
//
#include <hip/hip_runtime.h>
#include <hip/hip_bf16.h>

// ---------------------------------------------------------------------------
// PointNet++ VAE forward for MI455X (gfx1250, wave32, WMMA)
// Dense layers -> fused WMMA GEMM (v_wmma_f32_16x16x32_f16):
//   * weights pre-swizzled to WMMA B-fragment order -> 2x global_load_b128/lane
//   * A fragment is two contiguous 8-half runs       -> 2x global_load_b128/lane
//   * epilogue (bias + ReLU + inference-BN) templated -> branch-free
// ---------------------------------------------------------------------------

#define Bc   32
#define Nn   2048
#define N1c  410
#define N2c  103
#define Kc   64
#define LAT  20

typedef __attribute__((ext_vector_type(16))) _Float16 v16h;
typedef __attribute__((ext_vector_type(8)))  _Float16 v8h;
typedef __attribute__((ext_vector_type(8)))  float    v8f;

// ------------------------------ helpers ------------------------------------

__device__ __forceinline__ unsigned wang_hash(unsigned s) {
    s = (s ^ 61u) ^ (s >> 16); s *= 9u; s ^= s >> 4;
    s *= 0x27d4eb2du; s ^= s >> 15; return s;
}

__device__ __forceinline__ float gauss_hash(unsigned i) {
    unsigned a = wang_hash(i * 2654435761u + 1u);
    unsigned b = wang_hash(a ^ 0x9e3779b9u);
    float u1 = (a + 0.5f) * (1.0f / 4294967296.0f);
    float u2 = (b + 0.5f) * (1.0f / 4294967296.0f);
    return sqrtf(-2.0f * __logf(u1)) * __cosf(6.2831853071f * u2);
}

// ---------------- weight f32 -> f16, swizzled to B-fragment order -----------
// Fragment block for (n-tile tn, k-tile kt): 32 lanes x 16 halves contiguous.
// Element (lane l, e) = W[kt*32 + (e&7) + (l>>4)*8 + (e>>3)*16, tn*16 + (l&15)]  (K = (e&7) + hs*8 + (e>>3)*16)

__global__ void convert_w_swz_kernel(const float* __restrict__ W, _Float16* __restrict__ Wh,
                                     int din, int dout, int kpad) {
    int i = blockIdx.x * 256 + threadIdx.x;
    int total = kpad * dout;
    if (i >= total) return;
    int e   = i & 15;
    int l   = (i >> 4) & 31;
    int blk = i >> 9;                      // tn * (kpad/32) + kt
    int ktiles = kpad >> 5;
    int kt  = blk % ktiles;
    int tn  = blk / ktiles;
    int r = kt * 32 + (e & 7) + (l >> 4) * 8 + (e >> 3) * 16;
    int c = tn * 16 + (l & 15);
    Wh[i] = (r < din) ? (_Float16)W[(size_t)r * dout + c] : (_Float16)0.f;
}

// ------------------------------ FPS ----------------------------------------

__global__ void fps_kernel(const float* __restrict__ pos, int Np, int Nc,
                           float* __restrict__ ctr) {
    __shared__ float dist[2048];
    __shared__ float rv[256];
    __shared__ int   ri[256];
    int b = blockIdx.x, t = threadIdx.x;
    const float* P = pos + (size_t)b * Np * 3;
    float x0 = P[0], y0 = P[1], z0 = P[2];
    for (int i = t; i < Np; i += 256) {
        float dx = P[3*i] - x0, dy = P[3*i+1] - y0, dz = P[3*i+2] - z0;
        dist[i] = dx*dx + dy*dy + dz*dz;
    }
    if (t == 0) {
        float* o = ctr + (size_t)b * Nc * 3;
        o[0] = x0; o[1] = y0; o[2] = z0;
    }
    __syncthreads();
    for (int c = 1; c < Nc; ++c) {
        float bv = -1.f; int bi = 0;
        for (int i = t; i < Np; i += 256)
            if (dist[i] > bv) { bv = dist[i]; bi = i; }
        rv[t] = bv; ri[t] = bi;
        __syncthreads();
        for (int s = 128; s > 0; s >>= 1) {
            if (t < s && rv[t + s] > rv[t]) { rv[t] = rv[t + s]; ri[t] = ri[t + s]; }
            __syncthreads();
        }
        int nxt = ri[0];
        float px = P[3*nxt], py = P[3*nxt+1], pz = P[3*nxt+2];
        if (t == 0) {
            float* o = ctr + ((size_t)b * Nc + c) * 3;
            o[0] = px; o[1] = py; o[2] = pz;
        }
        for (int i = t; i < Np; i += 256) {
            float dx = P[3*i] - px, dy = P[3*i+1] - py, dz = P[3*i+2] - pz;
            float dn = dx*dx + dy*dy + dz*dz;
            if (dn < dist[i]) dist[i] = dn;
        }
        __syncthreads();
    }
}

// ------------------------- radius ball-query (K nearest) --------------------

__global__ void group_kernel(const float* __restrict__ pos, const float* __restrict__ ctr,
                             int Np, int Mc, float r2, int* __restrict__ nbr) {
    __shared__ float d2[2048];
    __shared__ float rv[256];
    __shared__ int   ri[256];
    int bm = blockIdx.x, t = threadIdx.x;
    int b = bm / Mc;
    const float* P = pos + (size_t)b * Np * 3;
    const float* C = ctr + (size_t)bm * 3;
    float cx = C[0], cy = C[1], cz = C[2];
    for (int i = t; i < Np; i += 256) {
        float dx = P[3*i] - cx, dy = P[3*i+1] - cy, dz = P[3*i+2] - cz;
        float d = dx*dx + dy*dy + dz*dz;
        d2[i] = (d <= r2) ? d : 3.0e38f;
    }
    __syncthreads();
    int* out = nbr + (size_t)bm * Kc;
    for (int k = 0; k < Kc; ++k) {
        float bv = 3.0e38f; int bi = -1;
        for (int i = t; i < Np; i += 256)
            if (d2[i] < bv) { bv = d2[i]; bi = i; }
        rv[t] = bv; ri[t] = bi;
        __syncthreads();
        for (int s = 128; s > 0; s >>= 1) {
            if (t < s && rv[t + s] < rv[t]) { rv[t] = rv[t + s]; ri[t] = ri[t + s]; }
            __syncthreads();
        }
        if (t == 0) {
            int sel = (rv[0] < 2.9e38f) ? ri[0] : -1;
            out[k] = sel;
            if (sel >= 0) d2[sel] = 3.0e38f;
        }
        __syncthreads();
    }
}

// --------------------------- message builders ------------------------------

__global__ void build_msg1_kernel(const float* __restrict__ pos, const float* __restrict__ ctr,
                                  const int* __restrict__ nbr, _Float16* __restrict__ msg) {
    int row = blockIdx.x * 256 + threadIdx.x;
    if (row >= Bc * N1c * Kc) return;
    int m = (row / Kc) % N1c;
    int b = row / (Kc * N1c);
    _Float16* o = msg + (size_t)row * 32;
    int j = nbr[row];
    float v[6] = {0.f, 0.f, 0.f, 0.f, 0.f, 0.f};
    if (j >= 0) {
        const float* p = pos + ((size_t)b * Nn + j) * 3;
        const float* c = ctr + ((size_t)b * N1c + m) * 3;
        v[0] = p[0]; v[1] = p[1]; v[2] = p[2];
        v[3] = p[0] - c[0]; v[4] = p[1] - c[1]; v[5] = p[2] - c[2];
    }
    for (int t2 = 0; t2 < 6; ++t2)  o[t2] = (_Float16)v[t2];
    for (int t2 = 6; t2 < 32; ++t2) o[t2] = (_Float16)0.f;
}

__global__ void build_msg2_kernel(const _Float16* __restrict__ f1, const float* __restrict__ p1,
                                  const float* __restrict__ ctr2, const int* __restrict__ nbr,
                                  _Float16* __restrict__ msg) {
    int row = blockIdx.x;                               // B*N2*K rows, 160 cols
    int t = threadIdx.x;                                // 64 threads
    int m = (row / Kc) % N2c;
    int b = row / (Kc * N2c);
    int j = nbr[row];
    _Float16* o = msg + (size_t)row * 160;
    if (j < 0) {
        for (int c = t; c < 160; c += 64) o[c] = (_Float16)0.f;
        return;
    }
    const _Float16* f = f1 + ((size_t)b * N1c + j) * 128;
    for (int c = t; c < 128; c += 64) o[c] = f[c];
    if (t < 32) {
        float v = 0.f;
        if (t < 3)
            v = p1[((size_t)b * N1c + j) * 3 + t] - ctr2[((size_t)b * N2c + m) * 3 + t];
        o[128 + t] = (_Float16)v;
    }
}

__global__ void build_msg3_kernel(const _Float16* __restrict__ f2, const float* __restrict__ ctr2,
                                  _Float16* __restrict__ msg) {
    int row = blockIdx.x;                               // B*N2 rows, 288 cols
    int t = threadIdx.x;                                // 64 threads
    const _Float16* f = f2 + (size_t)row * 256;
    _Float16* o = msg + (size_t)row * 288;
    for (int c = t; c < 256; c += 64) o[c] = f[c];
    if (t < 32) {
        float v = (t < 3) ? ctr2[(size_t)row * 3 + t] : 0.f;
        o[256 + t] = (_Float16)v;
    }
}

// -------------------------- fused WMMA GEMM --------------------------------
// A [M,Kpad] f16 row-major; Wf pre-swizzled fragment blocks of 512 halves per
// (n-tile, k-tile). Wave computes one 16x16 tile; 4 waves/block -> 64 columns.
// MODE 0: BN(ReLU(x+b))  MODE 1: ReLU(x+b)  MODE 2: x+b
// HALF_OUT: store f16 (next-layer operand) else f32.

template <int MODE, bool HALF_OUT>
__global__ void wmma_gemm_kernel(const _Float16* __restrict__ A,
                                 const _Float16* __restrict__ Wf,
                                 const float* __restrict__ bias,
                                 const float* __restrict__ gamma,
                                 const float* __restrict__ beta,
                                 void* __restrict__ outP,
                                 int M, int Kpad, int Nout) {
    int lane = threadIdx.x & 31;
    int wv   = threadIdx.x >> 5;
    int tm   = blockIdx.x;
    int tn   = blockIdx.y * 4 + wv;
    int hs   = lane >> 4;
    int lr   = lane & 15;
    int ktiles = Kpad >> 5;

    v8f acc = {0.f, 0.f, 0.f, 0.f, 0.f, 0.f, 0.f, 0.f};

    const _Float16* arow = A + (size_t)(tm * 16 + lr) * Kpad + hs * 8;
    const _Float16* wfr  = Wf + ((size_t)tn * ktiles * 32 + lane) * 16;

    for (int kt = 0; kt < ktiles; ++kt) {
        // A fragment: halves [hs*8 .. +8) and [16+hs*8 .. +8) of this 32-K slab
        v8h a0 = *(const v8h*)(arow);
        v8h a1 = *(const v8h*)(arow + 16);
        // B fragment: 32 contiguous bytes per lane from swizzled block
        v8h b0 = *(const v8h*)(wfr);
        v8h b1 = *(const v8h*)(wfr + 8);
        v16h a = __builtin_shufflevector(a0, a1, 0,1,2,3,4,5,6,7,8,9,10,11,12,13,14,15);
        v16h b = __builtin_shufflevector(b0, b1, 0,1,2,3,4,5,6,7,8,9,10,11,12,13,14,15);
        acc = __builtin_amdgcn_wmma_f32_16x16x32_f16(
            false, a, false, b, (short)0, acc, false, false);
        arow += 32;
        wfr  += 512;
    }

    const float inv = 0.9999975000f;   // 1/sqrt(1 + 5e-6)
    int n = tn * 16 + lr;
    float bn = bias[n];
    float gm = (MODE == 0) ? gamma[n] : 0.f;
    float bt = (MODE == 0) ? beta[n]  : 0.f;
    #pragma unroll
    for (int i = 0; i < 8; ++i) {
        int m = tm * 16 + i + hs * 8;
        float v = acc[i] + bn;
        if (MODE <= 1) v = fmaxf(v, 0.0f);
        if (MODE == 0) v = gm * v * inv + bt;
        if (HALF_OUT) ((_Float16*)outP)[(size_t)m * Nout + n] = (_Float16)v;
        else          ((float*)outP)[(size_t)m * Nout + n] = v;
    }
}

// ------------------------------ pooling ------------------------------------

__global__ void grouppool_kernel(const _Float16* __restrict__ h, const int* __restrict__ nbr,
                                 int C, _Float16* __restrict__ out) {
    int bm = blockIdx.x;          // B*Mc
    int c = threadIdx.x;          // blockDim.x == C
    const _Float16* hr = h + (size_t)bm * Kc * C;
    const int* nb = nbr + (size_t)bm * Kc;
    float best = -3.0e38f;
    for (int k = 0; k < Kc; ++k)
        if (nb[k] >= 0) best = fmaxf(best, (float)hr[(size_t)k * C + c]);
    out[(size_t)bm * C + c] = (_Float16)best;
}

__global__ void centerpool_kernel(const float* __restrict__ h, int Mc, int C,
                                  float* __restrict__ g) {
    int b = blockIdx.x, t = threadIdx.x;
    for (int c = t; c < C; c += blockDim.x) {
        float best = -3.0e38f;
        for (int m = 0; m < Mc; ++m)
            best = fmaxf(best, h[((size_t)b * Mc + m) * C + c]);
        g[(size_t)b * C + c] = best;
    }
}

// ------------------------------ VAE head -----------------------------------

__global__ void head_kernel(const float* __restrict__ g,
                            const float* __restrict__ muW, const float* __restrict__ muB,
                            const float* __restrict__ sgW, const float* __restrict__ sgB,
                            float* __restrict__ outMu, float* __restrict__ outLv,
                            _Float16* __restrict__ zpad) {
    int b = blockIdx.x, j = threadIdx.x;   // 32 threads
    zpad[b * 32 + j] = (_Float16)0.f;
    if (j >= LAT) return;
    const float* gb = g + (size_t)b * 512;
    float mu = muB[j], lv = sgB[j];
    for (int i = 0; i < 512; ++i) {
        float gv = gb[i];
        mu += gv * muW[i * LAT + j];
        lv += gv * sgW[i * LAT + j];
    }
    outMu[b * LAT + j] = mu;
    outLv[b * LAT + j] = lv;
    float eps = gauss_hash((unsigned)(b * LAT + j));
    float z = mu + eps * __expf(0.5f * lv);
    zpad[b * 32 + j] = (_Float16)z;
}

// ------------------------------ host side ----------------------------------

struct LayerW { int src; int din; int dout; int kpad; };
static const LayerW LWT[12] = {
    { 1,    6,   64,   32}, { 5,   64,   64,   64}, { 9,   64,  128,   64},
    {13,  131,  128,  160}, {17,  128,  128,  128}, {21,  128,  256,  128},
    {25,  259,  256,  288}, {29,  256,  256,  256}, {33,  256,  512,  256},
    {41,   20,  512,   32}, {43,  512, 1024,  512}, {45, 1024, 6144, 1024},
};

extern "C" void kernel_launch(void* const* d_in, const int* in_sizes, int n_in,
                              void* d_out, int out_size, void* d_ws, size_t ws_size,
                              hipStream_t stream) {
    (void)in_sizes; (void)n_in; (void)out_size; (void)ws_size;

    const float* x = (const float*)d_in[0];

    char* base = (char*)d_ws;
    size_t off = 0;
    auto alloc = [&](size_t bytes) -> void* {
        void* p = base + off;
        off += (bytes + 255) & ~(size_t)255;
        return p;
    };

    _Float16* wh[12];
    for (int i = 0; i < 12; ++i)
        wh[i] = (_Float16*)alloc((size_t)LWT[i].kpad * LWT[i].dout * sizeof(_Float16));

    const size_t M1 = (size_t)Bc * N1c * Kc;   // 839680
    const size_t M2 = (size_t)Bc * N2c * Kc;   // 210944
    const size_t M3 = (size_t)Bc * N2c;        // 3296

    float*    ctr1 = (float*)alloc((size_t)Bc * N1c * 3 * sizeof(float));
    float*    ctr2 = (float*)alloc((size_t)Bc * N2c * 3 * sizeof(float));
    int*      nbr1 = (int*)alloc(M1 * sizeof(int));
    int*      nbr2 = (int*)alloc(M2 * sizeof(int));
    _Float16* bufA = (_Float16*)alloc(M1 * 64  * sizeof(_Float16));
    _Float16* bufB = (_Float16*)alloc(M1 * 128 * sizeof(_Float16));
    _Float16* f1h  = (_Float16*)alloc((size_t)Bc * N1c * 128 * sizeof(_Float16));
    _Float16* f2h  = (_Float16*)alloc((size_t)Bc * N2c * 256 * sizeof(_Float16));
    float*    h3f  = (float*)alloc(M3 * 512 * sizeof(float));
    float*    gbuf = (float*)alloc((size_t)Bc * 512 * sizeof(float));
    _Float16* zpad = (_Float16*)alloc((size_t)Bc * 32 * sizeof(_Float16));
    _Float16* d1h  = (_Float16*)alloc((size_t)Bc * 512 * sizeof(_Float16));
    _Float16* d2h  = (_Float16*)alloc((size_t)Bc * 1024 * sizeof(_Float16));

    // weight conversion + fragment swizzle
    for (int i = 0; i < 12; ++i) {
        int total = LWT[i].kpad * LWT[i].dout;
        convert_w_swz_kernel<<<(total + 255) / 256, 256, 0, stream>>>(
            (const float*)d_in[LWT[i].src], wh[i], LWT[i].din, LWT[i].dout, LWT[i].kpad);
    }

    auto gemm = [&](const _Float16* Ab, int wi, void* outp, int M, int mode, bool halfOut,
                    const float* b_ovr = nullptr) {
        const float* bias = b_ovr ? b_ovr : (const float*)d_in[LWT[wi].src + 1];
        const float* gm   = (mode == 0) ? (const float*)d_in[LWT[wi].src + 2] : nullptr;
        const float* bt   = (mode == 0) ? (const float*)d_in[LWT[wi].src + 3] : nullptr;
        dim3 grid(M / 16, LWT[wi].dout / 64);
        int Kp = LWT[wi].kpad, No = LWT[wi].dout;
        if (mode == 0 && halfOut)
            wmma_gemm_kernel<0, true><<<grid, 128, 0, stream>>>(Ab, wh[wi], bias, gm, bt, outp, M, Kp, No);
        else if (mode == 0)
            wmma_gemm_kernel<0, false><<<grid, 128, 0, stream>>>(Ab, wh[wi], bias, gm, bt, outp, M, Kp, No);
        else if (mode == 1)
            wmma_gemm_kernel<1, true><<<grid, 128, 0, stream>>>(Ab, wh[wi], bias, gm, bt, outp, M, Kp, No);
        else
            wmma_gemm_kernel<2, false><<<grid, 128, 0, stream>>>(Ab, wh[wi], bias, gm, bt, outp, M, Kp, No);
    };

    // ---- SA1 ----
    fps_kernel<<<Bc, 256, 0, stream>>>(x, Nn, N1c, ctr1);
    group_kernel<<<Bc * N1c, 256, 0, stream>>>(x, ctr1, Nn, N1c, 0.04f, nbr1);
    build_msg1_kernel<<<(int)((M1 + 255) / 256), 256, 0, stream>>>(x, ctr1, nbr1, bufA);
    gemm(bufA, 0, bufB, (int)M1, 0, true);
    gemm(bufB, 1, bufA, (int)M1, 0, true);
    gemm(bufA, 2, bufB, (int)M1, 0, true);
    grouppool_kernel<<<Bc * N1c, 128, 0, stream>>>(bufB, nbr1, 128, f1h);

    // ---- SA2 ----
    fps_kernel<<<Bc, 256, 0, stream>>>(ctr1, N1c, N2c, ctr2);
    group_kernel<<<Bc * N2c, 256, 0, stream>>>(ctr1, ctr2, N1c, N2c, 0.16f, nbr2);
    build_msg2_kernel<<<(int)M2, 64, 0, stream>>>(f1h, ctr1, ctr2, nbr2, bufA);
    gemm(bufA, 3, bufB, (int)M2, 0, true);
    gemm(bufB, 4, bufA, (int)M2, 0, true);
    gemm(bufA, 5, bufB, (int)M2, 0, true);
    grouppool_kernel<<<Bc * N2c, 256, 0, stream>>>(bufB, nbr2, 256, f2h);

    // ---- SA3 (global) ----
    build_msg3_kernel<<<(int)M3, 64, 0, stream>>>(f2h, ctr2, bufA);
    gemm(bufA, 6, bufB, (int)M3, 0, true);
    gemm(bufB, 7, bufA, (int)M3, 0, true);
    gemm(bufA, 8, h3f, (int)M3, 0, false);
    centerpool_kernel<<<Bc, 256, 0, stream>>>(h3f, N2c, 512, gbuf);

    // ---- VAE head ----
    float* out_f  = (float*)d_out;
    float* out_mu = out_f + (size_t)Bc * Nn * 3;          // 196608
    float* out_lv = out_mu + (size_t)Bc * LAT;            // +640
    head_kernel<<<Bc, 32, 0, stream>>>(gbuf,
        (const float*)d_in[37], (const float*)d_in[38],
        (const float*)d_in[39], (const float*)d_in[40],
        out_mu, out_lv, zpad);

    // ---- decoder ----
    gemm(zpad, 9,  d1h, Bc, 1, true,  (const float*)d_in[42]);
    gemm(d1h,  10, d2h, Bc, 1, true,  (const float*)d_in[44]);
    gemm(d2h,  11, out_f, Bc, 2, false, (const float*)d_in[46]);
}
